// MultiHeadedAttention_52132313039168
// MI455X (gfx1250) — matrix operand order repaired and checked by reading, not compiled, not run on hardware
//
#include <hip/hip_runtime.h>
#include <stdint.h>

// Problem constants (from reference)
#define BB   8
#define NN   1024
#define EE   512
#define HH   8
#define DKK  64
#define KDIM 512
#define MROWS (BB * NN)   // 8192

typedef __attribute__((ext_vector_type(16))) __bf16 v16bf;
typedef __attribute__((ext_vector_type(8)))  float  v8f;
typedef __attribute__((ext_vector_type(4)))  int    v4i;

union FragAB {
    v16bf v;
    unsigned short u[16];
    uint4 q[2];
};

union Pack8 {
    unsigned short u[8];
    uint4 q;
};

__device__ __forceinline__ unsigned short f32_to_bf16(float f) {
    union { float f; uint32_t u; } c; c.f = f;
    uint32_t r = c.u + 0x7FFFu + ((c.u >> 16) & 1u);   // round-to-nearest-even
    return (unsigned short)(r >> 16);
}

// ---------------------------------------------------------------------------
// Global -> LDS staging: async DMA (ASYNCcnt) when the gfx1250 builtins are
// available, synchronous load+ds_store otherwise.
// Builtin signature (from clang diagnostic): (v4i AS1*, v4i AS3*, Ii, Ii).
// ---------------------------------------------------------------------------
#if defined(__has_builtin)
#if __has_builtin(__builtin_amdgcn_global_load_async_to_lds_b128) && \
    __has_builtin(__builtin_amdgcn_s_wait_asynccnt)
#define HAS_ASYNC_LDS 1
#endif
#endif

__device__ __forceinline__ void copy16_g2l(unsigned short* l,
                                           const unsigned short* g) {
#ifdef HAS_ASYNC_LDS
    typedef __attribute__((address_space(1))) v4i* gv4i_p;
    typedef __attribute__((address_space(3))) v4i* lv4i_p;
    __builtin_amdgcn_global_load_async_to_lds_b128(
        (gv4i_p)(g), (lv4i_p)(l), 0, 0);
#else
    *(uint4*)l = *(const uint4*)g;
#endif
}

__device__ __forceinline__ void wait_async_copies() {
#ifdef HAS_ASYNC_LDS
    __builtin_amdgcn_s_wait_asynccnt(0);
#endif
}

// ---------------------------------------------------------------------------
// fp32 -> bf16 conversion
// ---------------------------------------------------------------------------
__global__ void cvt_f32_bf16(const float* __restrict__ src,
                             unsigned short* __restrict__ dst, int n) {
    int i = blockIdx.x * blockDim.x + threadIdx.x;
    if (i < n) dst[i] = f32_to_bf16(src[i]);
}

// ---------------------------------------------------------------------------
// Y = X * W^T (+bias), X:[M,512] bf16 row-major, W:[512,512] bf16 row-major.
// Block: 32 X-rows staged in LDS (shared by 8 waves); each wave computes a
// 32(m) x 64(n) strip -> 8 v_wmma_f32_16x16x32_bf16 per 32-wide k-chunk.
// mode 0: store bf16 Q  [B,H,N,DK] scaled by 1/sqrt(DK)
// mode 1: store bf16 K  [B,H,N,DK]
// mode 2: store bf16 V^T[B,H,DK,N]
// mode 3: store f32  out[M,E] (+bias)
// ---------------------------------------------------------------------------
__global__ __launch_bounds__(256)
void gemm_bf16_wmma(const unsigned short* __restrict__ X,
                    const unsigned short* __restrict__ W,
                    const float* __restrict__ bias,
                    unsigned short* __restrict__ out_bf,
                    float* __restrict__ out_f32,
                    int mode, float scale)
{
    __shared__ unsigned short lsX[32 * KDIM];   // 32 KB

    const int wave = threadIdx.x >> 5;
    const int lane = threadIdx.x & 31;
    const int g    = lane >> 4;
    const int l16  = lane & 15;
    const int m0   = blockIdx.x * 32;
    const int n0   = wave * 64;

    // Cooperative stage of X[m0..m0+31][0..511] (32KB): 64 ushorts per thread.
    {
        const unsigned short* src = X + (size_t)m0 * KDIM + (size_t)threadIdx.x * 64;
        unsigned short* dst = lsX + threadIdx.x * 64;
#pragma unroll
        for (int j = 0; j < 8; ++j)
            copy16_g2l(dst + j * 8, src + j * 8);
        wait_async_copies();
    }
    __syncthreads();

    v8f acc[2][4] = {};
    for (int kb = 0; kb < KDIM; kb += 32) {
        FragAB a[2];
#pragma unroll
        for (int mt = 0; mt < 2; ++mt) {
            const unsigned short* ar = lsX + (mt * 16 + l16) * KDIM + kb;
            a[mt].q[0] = *(const uint4*)(ar + 16 * g);
            a[mt].q[1] = *(const uint4*)(ar + 16 * g + 8);
        }
        FragAB bf[4];
#pragma unroll
        for (int nt = 0; nt < 4; ++nt) {
            const unsigned short* wrow =
                W + (size_t)(n0 + nt * 16 + l16) * KDIM + kb + 16 * g;
            bf[nt].q[0] = *(const uint4*)(wrow);
            bf[nt].q[1] = *(const uint4*)(wrow + 8);
        }
#pragma unroll
        for (int nt = 0; nt < 4; ++nt)
#pragma unroll
            for (int mt = 0; mt < 2; ++mt)
                acc[mt][nt] = __builtin_amdgcn_wmma_f32_16x16x32_bf16(
                    false, a[mt].v, false, bf[nt].v, (short)0,
                    acc[mt][nt], false, false);
    }

#pragma unroll
    for (int nt = 0; nt < 4; ++nt) {
        const int n = n0 + nt * 16 + l16;
        const float bn = bias[n];
#pragma unroll
        for (int mt = 0; mt < 2; ++mt)
#pragma unroll
            for (int r = 0; r < 8; ++r) {
                const int m = m0 + mt * 16 + 8 * g + r;   // D row = r + 8g
                const float y = acc[mt][nt][r] + bn;
                if (mode == 3) {
                    out_f32[(size_t)m * EE + n] = y;
                } else {
                    const int b  = m >> 10, nq = m & (NN - 1);
                    const int h  = n >> 6,  d  = n & (DKK - 1);
                    const unsigned short v = f32_to_bf16(y * scale);
                    if (mode == 2)
                        out_bf[(((size_t)(b * HH + h)) * DKK + d) * NN + nq] = v;
                    else
                        out_bf[(((size_t)(b * HH + h)) * NN + nq) * DKK + d] = v;
                }
            }
    }
}

// ---------------------------------------------------------------------------
// Fused flash attention. 8 waves/block share one (b,h); each wave owns a
// 16-query tile. K and V^T chunks are double-buffered in LDS via async DMA.
// S^T = K*Q^T via WMMA; mask + online softmax (lane-local + shfl_xor 16);
// out^T += V^T * P^T via WMMA. Writes attn_out [B,N,E] bf16.
// ---------------------------------------------------------------------------
__global__ __launch_bounds__(256)
void attn_fused_wmma(const unsigned short* __restrict__ Qh,   // [B,H,N,DK] pre-scaled
                     const unsigned short* __restrict__ Kh,   // [B,H,N,DK]
                     const unsigned short* __restrict__ Vt,   // [B,H,DK,N]
                     const float* __restrict__ dist,          // [B,N,N]
                     const int*   __restrict__ pmask,         // [B,N,N]
                     unsigned short* __restrict__ attn_out)   // [B,N,E]
{
    __shared__ unsigned short lsK[2][32 * DKK];   // 2 x 4 KB: K rows kc..kc+31
    __shared__ unsigned short lsV[2][DKK * 32];   // 2 x 4 KB: V^T cols kc..kc+31

    const int wave = threadIdx.x >> 5;
    const int lane = threadIdx.x & 31;
    const int g    = lane >> 4;
    const int l16  = lane & 15;

    const int tid = blockIdx.x * 8 + wave;     // 0 .. B*H*64-1 (same bh per block)
    const int qt  = tid & 63;
    const int bh  = tid >> 6;
    const int h   = bh & (HH - 1);
    const int b   = bh >> 3;
    const int q0  = qt * 16;
    const int q   = q0 + l16;                  // this lane's query column
    const float thr = 0.2f + 0.1f * (float)h;  // DIST_BAR[h]

    const unsigned short* Qp = Qh + (size_t)bh * NN * DKK;
    const unsigned short* Kp = Kh + (size_t)bh * NN * DKK;
    const unsigned short* Vp = Vt + (size_t)bh * DKK * NN;
    const float* drow = dist  + ((size_t)b * NN + q) * NN;
    const int*   mrow = pmask + ((size_t)b * NN + q) * NN;

    // K chunk: 32 rows x 64 -> thread t copies 16B at (row=t>>3, d=(t&7)*8)
    // V chunk: 64 rows x 32 -> thread t copies 16B at (row=t>>2, kk=(t&3)*8)
    auto stage = [&](int buf, int kc) {
        const int t = threadIdx.x;
        copy16_g2l(&lsK[buf][(t >> 3) * DKK + (t & 7) * 8],
                   Kp + (size_t)(kc + (t >> 3)) * DKK + (t & 7) * 8);
        copy16_g2l(&lsV[buf][(t >> 2) * 32 + (t & 3) * 8],
                   Vp + (size_t)(t >> 2) * NN + kc + (t & 3) * 8);
    };

    // Preload Q B-fragments for the two 32-wide d-chunks
    FragAB bq[2];
    {
        const unsigned short* qrow = Qp + (size_t)q * DKK;
        bq[0].q[0] = *(const uint4*)(qrow + 16 * g);
        bq[0].q[1] = *(const uint4*)(qrow + 16 * g + 8);
        bq[1].q[0] = *(const uint4*)(qrow + 32 + 16 * g);
        bq[1].q[1] = *(const uint4*)(qrow + 32 + 16 * g + 8);
    }

    float m_i = -3.0e38f, l_i = 0.0f;
    v8f acc[4] = {};                           // out^T fragments, d-tiles 0..3

    stage(0, 0);
    wait_async_copies();
    __syncthreads();

    for (int c = 0; c < NN / 32; ++c) {
        const int kc  = c * 32;
        const int buf = c & 1;
        if (c + 1 < NN / 32) stage(buf ^ 1, kc + 32);

        const unsigned short* kbuf = lsK[buf];
        const unsigned short* vbuf = lsV[buf];

        // ---- S^T tiles: T0 (kk rel 0..15), T1 (kk rel 16..31) from LDS
        v8f st[2];
#pragma unroll
        for (int t = 0; t < 2; ++t) {
            const unsigned short* krow = kbuf + (t * 16 + l16) * DKK;
            FragAB a0, a1;
            a0.q[0] = *(const uint4*)(krow + 16 * g);
            a0.q[1] = *(const uint4*)(krow + 16 * g + 8);
            a1.q[0] = *(const uint4*)(krow + 32 + 16 * g);
            a1.q[1] = *(const uint4*)(krow + 32 + 16 * g + 8);
            v8f s = {};
            s = __builtin_amdgcn_wmma_f32_16x16x32_bf16(
                false, a0.v, false, bq[0].v, (short)0, s, false, false);
            s = __builtin_amdgcn_wmma_f32_16x16x32_bf16(
                false, a1.v, false, bq[1].v, (short)0, s, false, false);
            st[t] = s;
        }

        // ---- masking (pad mask, distance mask, global-atom row/col)
        float p[2][8];
        float cmax = -3.0e38f;
#pragma unroll
        for (int t = 0; t < 2; ++t) {
            const int kkb = kc + t * 16 + 8 * g;
            float4 d0 = *(const float4*)(drow + kkb);
            float4 d1 = *(const float4*)(drow + kkb + 4);
            int4  mm0 = *(const int4*)(mrow + kkb);
            int4  mm1 = *(const int4*)(mrow + kkb + 4);
            const float dv[8] = {d0.x, d0.y, d0.z, d0.w, d1.x, d1.y, d1.z, d1.w};
            const int   mv[8] = {mm0.x, mm0.y, mm0.z, mm0.w, mm1.x, mm1.y, mm1.z, mm1.w};
#pragma unroll
            for (int r = 0; r < 8; ++r) {
                const int kk = kkb + r;
                const bool allow =
                    (mv[r] != 0) && ((dv[r] < thr) || (q == 0) || (kk == 0));
                const float s = allow ? st[t][r] : -1.0e9f;
                p[t][r] = s;
                cmax = fmaxf(cmax, s);
            }
        }

        // ---- online softmax (per-query stats replicated in lanes l16 & l16+16)
        cmax = fmaxf(cmax, __shfl_xor(cmax, 16));
        const float m_new = fmaxf(m_i, cmax);
        const float alpha = __expf(m_i - m_new);
        float ls = 0.0f;
#pragma unroll
        for (int t = 0; t < 2; ++t)
#pragma unroll
            for (int r = 0; r < 8; ++r) {
                const float e = __expf(p[t][r] - m_new);
                p[t][r] = e;
                ls += e;
            }
        l_i = l_i * alpha + ls + __shfl_xor(ls, 16);
        m_i = m_new;
#pragma unroll
        for (int dt = 0; dt < 4; ++dt) acc[dt] *= alpha;

        // ---- build P^T B-fragment (32 kk x 16 q) from T0/T1 via lane exchange
        FragAB pb;
#pragma unroll
        for (int r = 0; r < 8; ++r) {
            const float o0 = p[0][r], o1 = p[1][r];
            const float x0 = __shfl_xor(o0, 16);
            const float x1 = __shfl_xor(o1, 16);
            pb.u[r]     = f32_to_bf16((g == 0) ? o0 : x1);  // kk rel 16g + r
            pb.u[r + 8] = f32_to_bf16((g == 0) ? x0 : o1);  // kk rel 16g + 8 + r
        }

        // ---- out^T += V^T * P^T for the four 16-wide d tiles (V^T from LDS)
#pragma unroll
        for (int dt = 0; dt < 4; ++dt) {
            const unsigned short* vrow = vbuf + (dt * 16 + l16) * 32;
            FragAB av;
            av.q[0] = *(const uint4*)(vrow + 16 * g);
            av.q[1] = *(const uint4*)(vrow + 16 * g + 8);
            acc[dt] = __builtin_amdgcn_wmma_f32_16x16x32_bf16(
                false, av.v, false, pb.v, (short)0, acc[dt], false, false);
        }

        wait_async_copies();   // next chunk's DMA must land before the barrier
        __syncthreads();
    }

    const float inv = (l_i > 0.0f) ? (1.0f / l_i) : 0.0f;
    unsigned short* orow = attn_out + ((size_t)b * NN + q) * EE + h * DKK;
#pragma unroll
    for (int dt = 0; dt < 4; ++dt) {
        Pack8 pk;
#pragma unroll
        for (int r = 0; r < 8; ++r)
            pk.u[r] = f32_to_bf16(acc[dt][r] * inv);      // d = dt*16 + 8g + r
        *(uint4*)(orow + dt * 16 + 8 * g) = pk.q;
    }
}

// ---------------------------------------------------------------------------
extern "C" void kernel_launch(void* const* d_in, const int* in_sizes, int n_in,
                              void* d_out, int out_size, void* d_ws, size_t ws_size,
                              hipStream_t stream) {
    (void)in_sizes; (void)n_in; (void)out_size; (void)ws_size;

    const float* query = (const float*)d_in[0];
    const float* keyi  = (const float*)d_in[1];
    const float* value = (const float*)d_in[2];
    const float* dist  = (const float*)d_in[3];
    const int*   maskp = (const int*)d_in[4];
    const float* Wq = (const float*)d_in[5];
    const float* bq = (const float*)d_in[6];
    const float* Wk = (const float*)d_in[7];
    const float* bk = (const float*)d_in[8];
    const float* Wv = (const float*)d_in[9];
    const float* bv = (const float*)d_in[10];
    const float* Wo = (const float*)d_in[11];
    const float* bo = (const float*)d_in[12];
    float* out = (float*)d_out;

    const size_t XE = (size_t)BB * NN * EE;   // 4,194,304 elements
    const size_t WE = (size_t)EE * EE;        // 262,144 elements
    unsigned short* ws = (unsigned short*)d_ws;
    unsigned short* xq = ws;
    unsigned short* xk = xq + XE;
    unsigned short* xv = xk + XE;
    unsigned short* wq = xv + XE;
    unsigned short* wk = wq + WE;
    unsigned short* wv = wk + WE;
    unsigned short* wo = wv + WE;
    unsigned short* Qh = wo + WE;
    unsigned short* Kh = Qh + XE;
    unsigned short* Vt = Kh + XE;
    unsigned short* at = Vt + XE;

    const int cb = 256;
    cvt_f32_bf16<<<(int)((XE + cb - 1) / cb), cb, 0, stream>>>(query, xq, (int)XE);
    cvt_f32_bf16<<<(int)((XE + cb - 1) / cb), cb, 0, stream>>>(keyi,  xk, (int)XE);
    cvt_f32_bf16<<<(int)((XE + cb - 1) / cb), cb, 0, stream>>>(value, xv, (int)XE);
    cvt_f32_bf16<<<(int)((WE + cb - 1) / cb), cb, 0, stream>>>(Wq, wq, (int)WE);
    cvt_f32_bf16<<<(int)((WE + cb - 1) / cb), cb, 0, stream>>>(Wk, wk, (int)WE);
    cvt_f32_bf16<<<(int)((WE + cb - 1) / cb), cb, 0, stream>>>(Wv, wv, (int)WE);
    cvt_f32_bf16<<<(int)((WE + cb - 1) / cb), cb, 0, stream>>>(Wo, wo, (int)WE);

    const dim3 gg(MROWS / 32);   // 256 blocks, 8 waves: one 32x64 strip per wave
    gemm_bf16_wmma<<<gg, 256, 0, stream>>>(xq, wq, bq, Qh, nullptr, 0, 0.125f); // 1/sqrt(64)
    gemm_bf16_wmma<<<gg, 256, 0, stream>>>(xk, wk, bk, Kh, nullptr, 1, 1.0f);
    gemm_bf16_wmma<<<gg, 256, 0, stream>>>(xv, wv, bv, Vt, nullptr, 2, 1.0f);

    attn_fused_wmma<<<dim3(BB * HH * (NN / 16) / 8), 256, 0, stream>>>(
        Qh, Kh, Vt, dist, maskp, at);

    gemm_bf16_wmma<<<gg, 256, 0, stream>>>(at, wo, bo, nullptr, out, 3, 1.0f);
}
